// SelfAttnPromptModel_2714419331461
// MI455X (gfx1250) — matrix operand order repaired and checked by reading, not compiled, not run on hardware
//
#include <hip/hip_runtime.h>
#include <hip/hip_bf16.h>

// ---------------- model constants ----------------
#define BB    4
#define DIM   1024
#define NHEAD 4
#define HDIM  256
#define NBLK  2
#define NPAD  3456            // 1025 + 2304 = 3329 -> pad to 27*128
#define MTOT  (BB * NPAD)     // 13824 rows, multiple of 128
#define MASKHW 48
#define MAXP  (MASKHW * MASKHW)

typedef __attribute__((ext_vector_type(16))) __bf16 v16bf;
typedef __attribute__((ext_vector_type(8)))  __bf16 bf8;
typedef __attribute__((ext_vector_type(8)))  float  v8f;
typedef __attribute__((ext_vector_type(4)))  unsigned int tdm_u4;
typedef __attribute__((ext_vector_type(8)))  int          tdm_i8;
typedef __attribute__((ext_vector_type(4)))  int          tdm_i4;

union FragBF { v16bf v; unsigned u[8]; };

__device__ __forceinline__ float gelu_exact(float x) {
    return 0.5f * x * (1.0f + erff(x * 0.70710678118654752f));
}

// --------- Tensor Data Mover: 2D tile load global->LDS (bf16 elements) ----------
// D# per CDNA5 ISA ch.8: group0 = {count/type/addr}, group1 = {sizes/strides/pad}.
// This toolchain exposes the 6-arg builtin: (g0 u32x4, g1 i32x8, g2 i32x4, g3 i32x4,
// g4 i32x8, cpol i32). Groups 2/3/4 unused for 2D tensors -> zeros.
__device__ __forceinline__ void tdm_load_2d_bf16(
    unsigned int lds_byte_addr, const void* gptr,
    unsigned int tensor_d0, unsigned int tensor_d1, unsigned long long stride0_elems,
    unsigned int tile_d0, unsigned int tile_d1,
    unsigned int pad_interval_code, unsigned int pad_amount_code, unsigned int pad_enable)
{
    unsigned long long ga = (unsigned long long)(size_t)gptr;
    tdm_u4 g0;
    g0[0] = 1u;                                    // count=1, user mode
    g0[1] = lds_byte_addr;                         // LDS destination (bytes)
    g0[2] = (unsigned int)ga;                      // global_addr[31:0]
    g0[3] = (unsigned int)(ga >> 32) | (2u << 30); // global_addr[56:32] | type=2
    tdm_i8 g1;
    g1[0] = (int)((1u << 16)                      // data_size = 1 -> 2 bytes
                | (pad_enable << 20)
                | (pad_interval_code << 22)
                | (pad_amount_code << 25));
    g1[1] = (int)((tensor_d0 & 0xFFFFu) << 16);                              // dim0[15:0]
    g1[2] = (int)((tensor_d0 >> 16) | ((tensor_d1 & 0xFFFFu) << 16));        // dim0[31:16]|dim1[15:0]
    g1[3] = (int)((tensor_d1 >> 16) | (tile_d0 << 16));                      // dim1[31:16]|tile0
    g1[4] = (int)(tile_d1 & 0xFFFFu);                                        // tile1 (tile2=0)
    g1[5] = (int)(unsigned int)stride0_elems;                                // stride0[31:0]
    g1[6] = (int)(unsigned int)(stride0_elems >> 32);                        // stride0[47:32]
    g1[7] = 0;
    tdm_i4 gz4 = {0, 0, 0, 0};
    tdm_i8 gz8 = {0, 0, 0, 0, 0, 0, 0, 0};
    __builtin_amdgcn_tensor_load_to_lds(g0, g1, gz4, gz4, gz8, 0);
}

// ---------------- small prep kernels ----------------

__global__ void f32_to_bf16_kernel(const float* __restrict__ s, __bf16* __restrict__ d, int n) {
    int i = blockIdx.x * 256 + threadIdx.x;
    if (i < n) d[i] = (__bf16)s[i];
}

// W[K][N] f32 -> Wt[N][K] bf16 (one-time, so TDM can stage B tiles without transpose)
__global__ void transpose_to_bf16_kernel(const float* __restrict__ W, __bf16* __restrict__ Wt,
                                         int K, int N) {
    long long idx = (long long)blockIdx.x * 256 + threadIdx.x;
    if (idx >= (long long)K * N) return;
    int k = (int)(idx / N), n = (int)(idx % N);
    Wt[(size_t)n * K + k] = (__bf16)W[idx];
}

// Compact prompt indices (row-major order == np.nonzero), per batch.
__global__ void prep_prompts_kernel(const int* __restrict__ mask, const float* __restrict__ depth,
                                    int* __restrict__ cnt, int* __restrict__ nvalid,
                                    float* __restrict__ px, float* __restrict__ py,
                                    float* __restrict__ pd) {
    int b = blockIdx.x;
    if (threadIdx.x != 0) return;
    int c = 0;
    for (int r = 0; r < MASKHW; ++r)
        for (int q = 0; q < MASKHW; ++q) {
            int v = mask[(b * MASKHW + r) * MASKHW + q];
            if (v > 0) {
                px[b * MAXP + c] = (q + 0.5f) / (float)MASKHW;
                py[b * MAXP + c] = (r + 0.5f) / (float)MASKHW;
                pd[b * MAXP + c] = depth[(b * MASKHW + r) * MASKHW + q];
                ++c;
            }
        }
    cnt[b] = c;
    nvalid[b] = 1025 + c;  // query(1024) + sep(1) + prompts
}

__global__ void query_pe_kernel(const float* __restrict__ gauss, float* __restrict__ qpe) {
    int n = blockIdx.x;
    float cx = 2.0f * (((n & 31) + 0.5f) / 32.0f) - 1.0f;
    float cy = 2.0f * (((n >> 5) + 0.5f) / 32.0f) - 1.0f;
    for (int k = threadIdx.x; k < DIM / 2; k += 256) {
        float t = 6.28318530717958648f * (cx * gauss[k] + cy * gauss[DIM / 2 + k]);
        qpe[(size_t)n * DIM + k]           = sinf(t);
        qpe[(size_t)n * DIM + DIM / 2 + k] = cosf(t);
    }
}

__global__ void zero_pad_kernel(float* __restrict__ X) {
    int r = 1024 + blockIdx.x, b = blockIdx.y;
    float* p = X + ((size_t)b * NPAD + r) * DIM;
    for (int c = threadIdx.x; c < DIM; c += 256) p[c] = 0.0f;
}

__global__ void assemble_query_kernel(const float* __restrict__ img, const float* __restrict__ qpe,
                                      float* __restrict__ X) {
    int n = blockIdx.x, b = blockIdx.y;
    for (int c = threadIdx.x; c < DIM; c += 256)
        X[((size_t)b * NPAD + n) * DIM + c] =
            img[((size_t)b * 1024 + n) * DIM + c] + qpe[(size_t)n * DIM + c];
}

// depth -> 512 hidden (gelu) -> 1024 token, + Fourier PE, written into X rows 1025+j
__global__ void prompt_tokens_kernel(const float* __restrict__ gauss,
                                     const float* __restrict__ w1, const float* __restrict__ b1,
                                     const float* __restrict__ w2, const float* __restrict__ b2,
                                     const int* __restrict__ cnt,
                                     const float* __restrict__ px, const float* __restrict__ py,
                                     const float* __restrict__ pd,
                                     float* __restrict__ X) {
    int j = blockIdx.x, b = blockIdx.y;
    if (j >= cnt[b]) return;  // uniform within block
    __shared__ float hid[DIM / 2];
    float d  = pd[b * MAXP + j];
    float cx = 2.0f * px[b * MAXP + j] - 1.0f;
    float cy = 2.0f * py[b * MAXP + j] - 1.0f;
    for (int k = threadIdx.x; k < DIM / 2; k += 256)
        hid[k] = gelu_exact(d * w1[k] + b1[k]);
    __syncthreads();
    for (int cc = 0; cc < 4; ++cc) {
        int c = threadIdx.x * 4 + cc;
        float acc = b2[c];
        for (int k = 0; k < DIM / 2; ++k) acc += hid[k] * w2[(size_t)k * DIM + c];
        int kk = (c < DIM / 2) ? c : c - DIM / 2;
        float t = 6.28318530717958648f * (cx * gauss[kk] + cy * gauss[DIM / 2 + kk]);
        float pe = (c < DIM / 2) ? sinf(t) : cosf(t);
        X[((size_t)b * NPAD + 1025 + j) * DIM + c] = acc + pe;
    }
}

__global__ void set_sep_kernel(float* __restrict__ X, const float* __restrict__ sep) {
    int b = blockIdx.x;
    for (int c = threadIdx.x; c < DIM; c += 256)
        X[((size_t)b * NPAD + 1024) * DIM + c] = sep[c];
}

__global__ __launch_bounds__(256) void layernorm_kernel(const float* __restrict__ X,
                                                        const float* __restrict__ w,
                                                        const float* __restrict__ bia,
                                                        __bf16* __restrict__ H) {
    size_t row = blockIdx.x;
    const float* x = X + row * DIM;
    __shared__ float rs[256], rs2[256];
    float vals[4], s = 0.0f, s2 = 0.0f;
#pragma unroll
    for (int i = 0; i < 4; ++i) {
        float v = x[threadIdx.x + i * 256];
        vals[i] = v; s += v; s2 += v * v;
    }
    rs[threadIdx.x] = s; rs2[threadIdx.x] = s2;
    __syncthreads();
    for (int o = 128; o > 0; o >>= 1) {
        if (threadIdx.x < o) { rs[threadIdx.x] += rs[threadIdx.x + o]; rs2[threadIdx.x] += rs2[threadIdx.x + o]; }
        __syncthreads();
    }
    float mean = rs[0] * (1.0f / DIM);
    float var  = rs2[0] * (1.0f / DIM) - mean * mean;
    float inv  = rsqrtf(var + 1e-5f);
#pragma unroll
    for (int i = 0; i < 4; ++i) {
        int c = threadIdx.x + i * 256;
        H[row * DIM + c] = (__bf16)((vals[i] - mean) * inv * w[c] + bia[c]);
    }
}

__global__ void extract_kernel(const float* __restrict__ X, float* __restrict__ out) {
    int n = blockIdx.x, b = blockIdx.y;
    for (int c = threadIdx.x; c < DIM; c += 256)
        out[((size_t)b * 1024 + n) * DIM + c] = X[((size_t)b * NPAD + n) * DIM + c];
}

// ---- bf16 WMMA GEMM with TDM double-buffered staging ----
// C[M,N] = A[M,K] * Bt[N,K]^T + bias (+gelu) (+residual).
// Block tile 128x128, 8 waves (4x2), wave tile 32x64 -> 8 wmma 16x16x32 per k-step.
// A/B tiles DMA'd by the Tensor Data Mover into LDS with 40-element padded rows.
#define GBM 128
#define GBN 128
#define GBK 32
#define T_ST 40   // padded LDS row stride (elements): 32 data + 8 pad

__global__ __launch_bounds__(256) void gemm_bf16_kernel(
    const __bf16* __restrict__ A, int lda,
    const __bf16* __restrict__ Bt, int ldbT,    // transposed weights [N][K]
    const float* __restrict__ bias,
    float* __restrict__ Cf, __bf16* __restrict__ Cb,
    const float* __restrict__ Res,
    int N, int K, int act_gelu)
{
    __shared__ __bf16 As[2][GBM * T_ST];
    __shared__ __bf16 Bs[2][GBN * T_ST];
    const int tid = threadIdx.x;
    const int wave = tid >> 5, lane = tid & 31;
    const int m15 = lane & 15, half = lane >> 4;
    const int wm = wave >> 1, wn = wave & 1;
    const int rowBase = blockIdx.y * GBM, colBase = blockIdx.x * GBN;
    const int nsteps = K / GBK;

    auto issue = [&](int kb, int buf) {
        // A tile: 128 rows x 32 k; pad 16B after each 64B row -> stride 40 elems
        tdm_load_2d_bf16((unsigned int)(size_t)&As[buf][0],
                         A + (size_t)rowBase * lda + kb,
                         (unsigned int)K, 1u << 20, (unsigned long long)lda,
                         GBK, GBM, 3u, 3u, 1u);
        // B tile: 128 n-rows x 32 k from transposed weights
        tdm_load_2d_bf16((unsigned int)(size_t)&Bs[buf][0],
                         Bt + (size_t)colBase * ldbT + kb,
                         (unsigned int)K, 1u << 20, (unsigned long long)ldbT,
                         GBK, GBN, 3u, 3u, 1u);
    };

    if (wave == 0) {
        issue(0, 0);
        __builtin_amdgcn_s_wait_tensorcnt(0);
    }
    __syncthreads();

    v8f acc[2][4] = {};
    int cur = 0;
    for (int step = 0; step < nsteps; ++step) {
        if (wave == 0 && step + 1 < nsteps) issue((step + 1) * GBK, cur ^ 1);

        FragBF a[2], bf[4];
#pragma unroll
        for (int mi = 0; mi < 2; ++mi) {
            int R = wm * 32 + mi * 16 + m15;
#pragma unroll
            for (int v = 0; v < 8; ++v) {
                int kk = ((v < 4) ? 2 * v : 16 + 2 * (v - 4)) + half * 8;  // ISA 16-bit A layout
                a[mi].u[v] = *(const unsigned*)&As[cur][R * T_ST + kk];
            }
        }
#pragma unroll
        for (int ni = 0; ni < 4; ++ni) {
            int C0 = wn * 64 + ni * 16 + m15;
#pragma unroll
            for (int v = 0; v < 8; ++v) {
                int kk = ((v < 4) ? 2 * v : 16 + 2 * (v - 4)) + half * 8;  // ISA 16-bit B layout
                bf[ni].u[v] = *(const unsigned*)&Bs[cur][C0 * T_ST + kk];
            }
        }
#pragma unroll
        for (int mi = 0; mi < 2; ++mi)
#pragma unroll
            for (int ni = 0; ni < 4; ++ni)
                acc[mi][ni] = __builtin_amdgcn_wmma_f32_16x16x32_bf16(
                    false, a[mi].v, false, bf[ni].v, (short)0, acc[mi][ni], false, false);

        __syncthreads();                                   // all done reading buffers[cur]
        if (wave == 0 && step + 1 < nsteps)
            __builtin_amdgcn_s_wait_tensorcnt(0);          // next tile landed
        __syncthreads();                                   // publish
        cur ^= 1;
    }

    // epilogue
#pragma unroll
    for (int mi = 0; mi < 2; ++mi)
#pragma unroll
        for (int ni = 0; ni < 4; ++ni)
#pragma unroll
            for (int v = 0; v < 8; ++v) {
                int row = rowBase + wm * 32 + mi * 16 + v + half * 8;  // C layout
                int col = colBase + wn * 64 + ni * 16 + m15;
                float x = acc[mi][ni][v] + bias[col];
                if (act_gelu) x = gelu_exact(x);
                size_t idx = (size_t)row * N + col;
                if (Res) x += Res[idx];
                if (Cf) Cf[idx] = x;
                if (Cb) Cb[idx] = (__bf16)x;
            }
}

// ---------------- flash attention (bf16 WMMA, f32 online softmax) ----------------
// Grid: (NPAD/128, NHEAD, B). 8 waves/block, each wave owns a 16-row query tile.
// K tile staged by TDM (wave 0) while all waves cooperatively transpose V into LDS.
#define K_STRIDE 264
#define V_STRIDE 40

__global__ __launch_bounds__(256) void attention_kernel(
    const __bf16* __restrict__ QKV, __bf16* __restrict__ O, const int* __restrict__ nvalid)
{
    __shared__ __bf16 Klds[32 * K_STRIDE];     // [key][feat], TDM-padded rows
    __shared__ __bf16 Vt[HDIM * V_STRIDE];     // transposed: [feat][key]
    __shared__ __bf16 Plds[8 * 16 * 32];       // per-wave P tile (16x32)
    const int b = blockIdx.z, h = blockIdx.y;
    const int tid = threadIdx.x, wave = tid >> 5, lane = tid & 31;
    const int m15 = lane & 15, half = lane >> 4;
    const int qrow0 = blockIdx.x * 128 + wave * 16;
    const int nv = nvalid[b];

    const __bf16* qptr  = QKV + ((size_t)b * NPAD + qrow0) * (3 * DIM) + (size_t)h * HDIM;
    const __bf16* kbase = QKV + (size_t)b * NPAD * (3 * DIM) + DIM     + (size_t)h * HDIM;
    const __bf16* vbase = QKV + (size_t)b * NPAD * (3 * DIM) + 2 * DIM + (size_t)h * HDIM;

    // Q fragments: 8 k-chunks of 32 features (A layout, read once)
    FragBF qf[8];
#pragma unroll
    for (int c = 0; c < 8; ++c)
#pragma unroll
        for (int v = 0; v < 8; ++v) {
            int kk = c * 32 + ((v < 4) ? 2 * v : 16 + 2 * (v - 4)) + half * 8;
            qf[c].u[v] = *(const unsigned*)(qptr + (size_t)m15 * (3 * DIM) + kk);
        }

    float mrun[8], lrun[8];
#pragma unroll
    for (int v = 0; v < 8; ++v) { mrun[v] = -1e30f; lrun[v] = 0.0f; }
    v8f o[16] = {};

    for (int kb = 0; kb < nv; kb += 32) {
        __syncthreads();
        if (wave == 0) {
            // K tile: 32 keys x 256 feats; pad 16B after each 512B row -> stride 264
            tdm_load_2d_bf16((unsigned int)(size_t)&Klds[0],
                             kbase + (size_t)kb * (3 * DIM),
                             HDIM, 1u << 20, (unsigned long long)(3 * DIM),
                             HDIM, 32, 6u, 3u, 1u);
        }
        {   // cooperative V transpose staging (all waves, overlapped with TDM)
            int kr = tid >> 3, grp = tid & 7;
            const __bf16* vs = vbase + (size_t)(kb + kr) * (3 * DIM) + grp * 32;
#pragma unroll
            for (int i = 0; i < 32; ++i) Vt[(grp * 32 + i) * V_STRIDE + kr] = vs[i];
        }
        if (wave == 0) __builtin_amdgcn_s_wait_tensorcnt(0);
        __syncthreads();

        // S = Q * K^T : two 16x16 tiles over the 32-key chunk, 8 wmma each (K=256 feats)
        v8f s[2] = {};
#pragma unroll
        for (int st = 0; st < 2; ++st)
#pragma unroll
            for (int c = 0; c < 8; ++c) {
                FragBF kf;
#pragma unroll
                for (int v = 0; v < 8; ++v) {
                    int kk = c * 32 + ((v < 4) ? 2 * v : 16 + 2 * (v - 4)) + half * 8;
                    kf.u[v] = *(const unsigned*)&Klds[(st * 16 + m15) * K_STRIDE + kk];
                }
                s[st] = __builtin_amdgcn_wmma_f32_16x16x32_bf16(
                    false, qf[c].v, false, kf.v, (short)0, s[st], false, false);
            }
        // scale 1/sqrt(256) and mask padded keys
#pragma unroll
        for (int st = 0; st < 2; ++st) {
            bool valid = (kb + st * 16 + m15) < nv;
#pragma unroll
            for (int v = 0; v < 8; ++v)
                s[st][v] = valid ? s[st][v] * 0.0625f : -1e30f;
        }
        // online softmax; row r lives in v=(r&7), lanes (r>>3)*16..+15 => 16-lane reductions
        float alpha[8];
        v8f p[2];
#pragma unroll
        for (int v = 0; v < 8; ++v) {
            float rm = fmaxf(s[0][v], s[1][v]);
            for (int off = 1; off < 16; off <<= 1) rm = fmaxf(rm, __shfl_xor(rm, off, 32));
            float nm = fmaxf(mrun[v], rm);
            alpha[v] = __expf(mrun[v] - nm);
            mrun[v] = nm;
            float p0 = __expf(s[0][v] - nm), p1 = __expf(s[1][v] - nm);
            p[0][v] = p0; p[1][v] = p1;
            float rs = p0 + p1;
            for (int off = 1; off < 16; off <<= 1) rs += __shfl_xor(rs, off, 32);
            lrun[v] = lrun[v] * alpha[v] + rs;
        }
#pragma unroll
        for (int t = 0; t < 16; ++t)
#pragma unroll
            for (int v = 0; v < 8; ++v) o[t][v] *= alpha[v];

        // C-layout -> A-layout relayout of P through LDS
        __bf16* pw = Plds + wave * 512;
#pragma unroll
        for (int v = 0; v < 8; ++v) {
            int row = v + half * 8;
            pw[row * 32 + m15]      = (__bf16)p[0][v];
            pw[row * 32 + 16 + m15] = (__bf16)p[1][v];
        }
        __syncthreads();
        FragBF pf;
#pragma unroll
        for (int v = 0; v < 8; ++v) {
            int kk = ((v < 4) ? 2 * v : 16 + 2 * (v - 4)) + half * 8;
            pf.u[v] = *(const unsigned*)&pw[m15 * 32 + kk];
        }
        // O += P * V : 16 feature tiles
#pragma unroll
        for (int t = 0; t < 16; ++t) {
            FragBF vf;
#pragma unroll
            for (int v = 0; v < 8; ++v) {
                int kk = ((v < 4) ? 2 * v : 16 + 2 * (v - 4)) + half * 8;
                vf.u[v] = *(const unsigned*)&Vt[(t * 16 + m15) * V_STRIDE + kk];
            }
            o[t] = __builtin_amdgcn_wmma_f32_16x16x32_bf16(
                false, pf.v, false, vf.v, (short)0, o[t], false, false);
        }
    }

    float inv[8];
#pragma unroll
    for (int v = 0; v < 8; ++v) inv[v] = (lrun[v] > 0.0f) ? 1.0f / lrun[v] : 0.0f;
    __bf16* optr = O + ((size_t)b * NPAD + qrow0) * DIM + (size_t)h * HDIM;
#pragma unroll
    for (int t = 0; t < 16; ++t)
#pragma unroll
        for (int v = 0; v < 8; ++v)
            optr[(size_t)(v + half * 8) * DIM + t * 16 + m15] = (__bf16)(o[t][v] * inv[v]);
}

// ---------------- host launch ----------------

extern "C" void kernel_launch(void* const* d_in, const int* in_sizes, int n_in,
                              void* d_out, int out_size, void* d_ws, size_t ws_size,
                              hipStream_t stream) {
    const float* img    = (const float*)d_in[0];
    const float* pdepth = (const float*)d_in[1];
    const int*   pmask  = (const int*)d_in[2];
    const float* gauss  = (const float*)d_in[5];
    const float* d2f_w1 = (const float*)d_in[6];
    const float* d2f_b1 = (const float*)d_in[7];
    const float* d2f_w2 = (const float*)d_in[8];
    const float* d2f_b2 = (const float*)d_in[9];
    const float* sep    = (const float*)d_in[10];
    const float* ln1_w  = (const float*)d_in[11];
    const float* ln1_b  = (const float*)d_in[12];
    const float* qkv_w  = (const float*)d_in[13];
    const float* qkv_b  = (const float*)d_in[14];
    const float* proj_w = (const float*)d_in[15];
    const float* proj_b = (const float*)d_in[16];
    const float* ln2_w  = (const float*)d_in[17];
    const float* ln2_b  = (const float*)d_in[18];
    const float* mlp_w1 = (const float*)d_in[19];
    const float* mlp_b1 = (const float*)d_in[20];
    const float* mlp_w2 = (const float*)d_in[21];
    const float* mlp_b2 = (const float*)d_in[22];

    // workspace arena
    char* w = (char*)d_ws;
    auto alloc = [&](size_t bytes) { char* p = w; w += (bytes + 255) & ~(size_t)255; return p; };
    int*    cnt    = (int*)   alloc(BB * sizeof(int));
    int*    nvalid = (int*)   alloc(BB * sizeof(int));
    float*  px     = (float*) alloc((size_t)BB * MAXP * 4);
    float*  py     = (float*) alloc((size_t)BB * MAXP * 4);
    float*  pd     = (float*) alloc((size_t)BB * MAXP * 4);
    float*  qpe    = (float*) alloc((size_t)1024 * DIM * 4);
    float*  X      = (float*) alloc((size_t)MTOT * DIM * 4);
    __bf16* H      = (__bf16*)alloc((size_t)MTOT * DIM * 2);
    __bf16* QKVb   = (__bf16*)alloc((size_t)MTOT * 3 * DIM * 2);
    __bf16* Ob     = (__bf16*)alloc((size_t)MTOT * DIM * 2);
    __bf16* HID    = (__bf16*)alloc((size_t)MTOT * 4 * DIM * 2);
    __bf16* WqkvT  = (__bf16*)alloc((size_t)NBLK * DIM * 3 * DIM * 2);   // [3072][1024] per blk
    __bf16* WprojT = (__bf16*)alloc((size_t)NBLK * DIM * DIM * 2);       // [1024][1024]
    __bf16* W1T    = (__bf16*)alloc((size_t)NBLK * DIM * 4 * DIM * 2);   // [4096][1024]
    __bf16* W2T    = (__bf16*)alloc((size_t)NBLK * 4 * DIM * DIM * 2);   // [1024][4096]
    (void)ws_size; (void)in_sizes; (void)n_in; (void)out_size;
    (void)f32_to_bf16_kernel;

    auto trp = [&](const float* s, __bf16* d, int K, int N) {
        long long tot = (long long)K * N;
        transpose_to_bf16_kernel<<<(unsigned)((tot + 255) / 256), 256, 0, stream>>>(s, d, K, N);
    };
    for (int i = 0; i < NBLK; ++i) {
        trp(qkv_w  + (size_t)i * DIM * 3 * DIM, WqkvT  + (size_t)i * 3 * DIM * DIM, DIM, 3 * DIM);
        trp(proj_w + (size_t)i * DIM * DIM,     WprojT + (size_t)i * DIM * DIM,     DIM, DIM);
        trp(mlp_w1 + (size_t)i * DIM * 4 * DIM, W1T    + (size_t)i * 4 * DIM * DIM, DIM, 4 * DIM);
        trp(mlp_w2 + (size_t)i * 4 * DIM * DIM, W2T    + (size_t)i * DIM * 4 * DIM, 4 * DIM, DIM);
    }

    prep_prompts_kernel<<<BB, 32, 0, stream>>>(pmask, pdepth, cnt, nvalid, px, py, pd);
    query_pe_kernel<<<1024, 256, 0, stream>>>(gauss, qpe);
    zero_pad_kernel<<<dim3(NPAD - 1024, BB), 256, 0, stream>>>(X);
    assemble_query_kernel<<<dim3(1024, BB), 256, 0, stream>>>(img, qpe, X);
    prompt_tokens_kernel<<<dim3(MAXP, BB), 256, 0, stream>>>(gauss, d2f_w1, d2f_b1, d2f_w2,
                                                             d2f_b2, cnt, px, py, pd, X);

    for (int i = 0; i < NBLK; ++i) {
        set_sep_kernel<<<BB, 256, 0, stream>>>(X, sep + (size_t)i * DIM);
        layernorm_kernel<<<MTOT, 256, 0, stream>>>(X, ln1_w + i * DIM, ln1_b + i * DIM, H);
        gemm_bf16_kernel<<<dim3(3 * DIM / GBN, MTOT / GBM), 256, 0, stream>>>(
            H, DIM, WqkvT + (size_t)i * 3 * DIM * DIM, DIM, qkv_b + i * 3 * DIM,
            nullptr, QKVb, nullptr, 3 * DIM, DIM, 0);
        attention_kernel<<<dim3(NPAD / 128, NHEAD, BB), 256, 0, stream>>>(QKVb, Ob, nvalid);
        gemm_bf16_kernel<<<dim3(DIM / GBN, MTOT / GBM), 256, 0, stream>>>(
            Ob, DIM, WprojT + (size_t)i * DIM * DIM, DIM, proj_b + i * DIM,
            X, nullptr, X, DIM, DIM, 0);
        layernorm_kernel<<<MTOT, 256, 0, stream>>>(X, ln2_w + i * DIM, ln2_b + i * DIM, H);
        gemm_bf16_kernel<<<dim3(4 * DIM / GBN, MTOT / GBM), 256, 0, stream>>>(
            H, DIM, W1T + (size_t)i * 4 * DIM * DIM, DIM, mlp_b1 + i * 4 * DIM,
            nullptr, HID, nullptr, 4 * DIM, DIM, 1);
        gemm_bf16_kernel<<<dim3(DIM / GBN, MTOT / GBM), 256, 0, stream>>>(
            HID, 4 * DIM, W2T + (size_t)i * DIM * 4 * DIM, 4 * DIM, mlp_b2 + i * DIM,
            X, nullptr, X, DIM, 4 * DIM, 0);
    }
    extract_kernel<<<dim3(1024, BB), 256, 0, stream>>>(X, (float*)d_out);
}
